// SRU_Formula_Cell_71631464563051
// MI455X (gfx1250) — matrix operand order repaired and checked by reading, not compiled, not run on hardware
//
#include <hip/hip_runtime.h>

typedef __bf16 bf16;
typedef bf16  v8bf  __attribute__((ext_vector_type(8)));
typedef bf16  v16bf __attribute__((ext_vector_type(16)));
typedef float v8f   __attribute__((ext_vector_type(8)));

#define T_STEPS 128
#define BATCH   64
#define NIN     1024
#define NOUT    2048
#define EMBED   1024
#define MROWS   (T_STEPS * BATCH)      // 8192
#define BN      (BATCH * NOUT)         // 131072

// ---------------------------------------------------------------------------
// f32 -> bf16 elementwise convert
// ---------------------------------------------------------------------------
__global__ __launch_bounds__(256) void cvt_f32_bf16(const float* __restrict__ x,
                                                    bf16* __restrict__ y, int n) {
  int i = blockIdx.x * 256 + threadIdx.x;
  if (i < n) y[i] = (bf16)x[i];
}

// ---------------------------------------------------------------------------
// [K,N] f32 -> [N,K] bf16 tiled transpose (K,N multiples of 32)
// out[n][k] = (bf16) in[k][n]
// ---------------------------------------------------------------------------
__global__ __launch_bounds__(256) void transpose_cvt(const float* __restrict__ in,
                                                     bf16* __restrict__ out,
                                                     int K, int N) {
  __shared__ bf16 tile[32][33];
  int n0 = blockIdx.x * 32, k0 = blockIdx.y * 32;
  int tx = threadIdx.x & 31, ty = threadIdx.x >> 5;   // 32 x 8 threads
#pragma unroll
  for (int i = 0; i < 32; i += 8)
    tile[ty + i][tx] = (bf16)in[(size_t)(k0 + ty + i) * N + (n0 + tx)];
  __syncthreads();
#pragma unroll
  for (int i = 0; i < 32; i += 8)
    out[(size_t)(n0 + ty + i) * K + (k0 + tx)] = tile[tx][ty + i];
}

// ---------------------------------------------------------------------------
// WMMA GEMM: out[M,N] = act( A[M,K](bf16) x Bt[N,K](bf16)^T + bias[N] )
// Block = 256 threads = 8 waves arranged 4(M) x 2(N): macro tile 128M x 64N.
// Each wave: 32M x 32N slab = 2x2 register tile (4 accumulators); each A and
// B fragment is reused twice in registers -> 4 WMMAs per 128B/lane loaded.
// act: 0 = identity, 1 = sigmoid
// ---------------------------------------------------------------------------
__device__ __forceinline__ v8f wmma_bf16(v16bf a, v16bf b, v8f c) {
  return __builtin_amdgcn_wmma_f32_16x16x32_bf16(false, a, false, b,
                                                 (short)0, c, false, false);
}

__device__ __forceinline__ v16bf load_afrag(const bf16* p) {
  v8bf lo = *(const v8bf*)(p);
  v8bf hi = *(const v8bf*)(p + 16);
  return __builtin_shufflevector(lo, hi, 0, 1, 2, 3, 4, 5, 6, 7,
                                 8, 9, 10, 11, 12, 13, 14, 15);
}

__global__ __launch_bounds__(256) void wmma_gemm(const bf16* __restrict__ A,
                                                 const bf16* __restrict__ Bt,
                                                 const float* __restrict__ bias,
                                                 float* __restrict__ out,
                                                 int M, int N, int K, int act) {
  const int wave = threadIdx.x >> 5;
  const int lane = threadIdx.x & 31;
  const int half = lane >> 4;       // K sub-chunk selector (fragment layout)
  const int r    = lane & 15;       // row (A) / column (B,C,D) within 16-tile
  const int wm   = wave >> 1;       // 0..3
  const int wn   = wave & 1;        // 0..1
  const int m0 = blockIdx.x * 128 + wm * 32;
  const int n0 = blockIdx.y * 64 + wn * 32;
  const int nA = n0 + r;
  const int nB = n0 + 16 + r;

  float bA = bias ? bias[nA] : 0.0f;
  float bB = bias ? bias[nB] : 0.0f;
  v8f acc00, acc01, acc10, acc11;
#pragma unroll
  for (int v = 0; v < 8; ++v) {
    acc00[v] = bA; acc10[v] = bA;
    acc01[v] = bB; acc11[v] = bB;
  }

  // 16-bit A fragment: lanes 0-15 hold K {0..7,16..23} of row r, lanes 16-31
  // hold K {8..15,24..31} -> two contiguous 16B chunks per lane.
  const bf16* ap0 = A  + (size_t)(m0 + r) * K + half * 8;
  const bf16* ap1 = A  + (size_t)(m0 + 16 + r) * K + half * 8;
  // 16-bit B fragment: same K layout as A (lanes 0-15 K {0..7,16..23}, lanes
  // 16-31 K {8..15,24..31}) -> two contiguous 16B chunks per lane from Bt[N,K].
  const bf16* bp0 = Bt + (size_t)nA * K + half * 8;
  const bf16* bp1 = Bt + (size_t)nB * K + half * 8;

  for (int k0 = 0; k0 < K; k0 += 32) {
    // Unguarded WGP-scope prefetch of the panels ~8 k-tiles ahead
    // (locality 3 -> near scope; slight tail over-prefetch stays inside
    //  mapped workspace and is harmless).
    __builtin_prefetch(ap0 + k0 + 256, 0, 3);
    __builtin_prefetch(ap1 + k0 + 256, 0, 3);
    __builtin_prefetch(bp0 + k0 + 256, 0, 3);
    __builtin_prefetch(bp1 + k0 + 256, 0, 3);

    v16bf a0 = load_afrag(ap0 + k0);
    v16bf a1 = load_afrag(ap1 + k0);
    v16bf b0 = load_afrag(bp0 + k0);
    v16bf b1 = load_afrag(bp1 + k0);
    acc00 = wmma_bf16(a0, b0, acc00);
    acc01 = wmma_bf16(a0, b1, acc01);
    acc10 = wmma_bf16(a1, b0, acc10);
    acc11 = wmma_bf16(a1, b1, acc11);
  }

  // C/D layout: lanes 0-15 -> M = v, lanes 16-31 -> M = 8+v; N = lane&15.
#pragma unroll
  for (int v = 0; v < 8; ++v) {
    int mA = m0 + half * 8 + v;
    int mB = m0 + 16 + half * 8 + v;
    float y00 = acc00[v], y01 = acc01[v], y10 = acc10[v], y11 = acc11[v];
    if (act == 1) {
      y00 = 1.0f / (1.0f + __expf(-y00));
      y01 = 1.0f / (1.0f + __expf(-y01));
      y10 = 1.0f / (1.0f + __expf(-y10));
      y11 = 1.0f / (1.0f + __expf(-y11));
    }
    out[(size_t)mA * N + nA] = y00;
    out[(size_t)mA * N + nB] = y01;
    out[(size_t)mB * N + nA] = y10;
    out[(size_t)mB * N + nB] = y11;
  }
}

// ---------------------------------------------------------------------------
// SRU recurrence + output, fused:
//   c_t = f_t * c_{t-1} + (1-f_t) * xp_t
//   h_t = r_t * tanh(c_t) + (1-r_t) * cx_t
// One thread per (b, n); sequential loop over T with coalesced slabs.
// ---------------------------------------------------------------------------
__global__ __launch_bounds__(256) void sru_scan(const float* __restrict__ f,
                                                const float* __restrict__ xp,
                                                const float* __restrict__ r,
                                                const float* __restrict__ cx,
                                                const float* __restrict__ c0,
                                                float* __restrict__ h,
                                                float* __restrict__ c_last) {
  int j = blockIdx.x * 256 + threadIdx.x;
  if (j >= BN) return;
  float c = c0[j];
  for (int t = 0; t < T_STEPS; ++t) {
    size_t off = (size_t)t * BN + j;
    float ft = f[off];
    c = ft * c + (1.0f - ft) * xp[off];
    float rt = r[off];
    h[off] = rt * tanhf(c) + (1.0f - rt) * cx[off];
  }
  if (c_last) c_last[j] = c;
}

// ---------------------------------------------------------------------------
// Host-side orchestration
// ---------------------------------------------------------------------------
extern "C" void kernel_launch(void* const* d_in, const int* in_sizes, int n_in,
                              void* d_out, int out_size, void* d_ws, size_t ws_size,
                              hipStream_t stream) {
  const float* xt  = (const float*)d_in[0];   // [T,B,NIN]
  const float* ct  = (const float*)d_in[1];   // [LAYERS,B,NOUT]
  const float* Wx  = (const float*)d_in[2];   // [NIN,NOUT]
  const float* Wf  = (const float*)d_in[3];
  const float* bf_ = (const float*)d_in[4];   // [NOUT]
  const float* Wr  = (const float*)d_in[5];
  const float* br  = (const float*)d_in[6];
  const float* Wcx = (const float*)d_in[7];
  const float* bcx = (const float*)d_in[8];
  const float* Wcl = (const float*)d_in[9];   // [NOUT,EMBED]
  const float* bcl = (const float*)d_in[10];  // [EMBED]
  float* out = (float*)d_out;

  // Workspace layout (bytes)
  char* ws = (char*)d_ws;
  size_t o = 0;
  bf16* WxT  = (bf16*)(ws + o); o += (size_t)NOUT * NIN * 2;    // [NOUT,NIN]
  bf16* WfT  = (bf16*)(ws + o); o += (size_t)NOUT * NIN * 2;
  bf16* WrT  = (bf16*)(ws + o); o += (size_t)NOUT * NIN * 2;
  bf16* WcxT = (bf16*)(ws + o); o += (size_t)NOUT * NIN * 2;
  bf16* WclT = (bf16*)(ws + o); o += (size_t)EMBED * NOUT * 2;  // [EMBED,NOUT]
  bf16* Xbf  = (bf16*)(ws + o); o += (size_t)MROWS * NIN * 2;   // layer-1 A
  bf16* Hbf  = (bf16*)(ws + o); o += (size_t)MROWS * NOUT * 2;  // h1 as A
  bf16* X2bf = (bf16*)(ws + o); o += (size_t)MROWS * NIN * 2;   // layer-2 A
  float* xp  = (float*)(ws + o); o += (size_t)MROWS * NOUT * 4;
  float* fg  = (float*)(ws + o); o += (size_t)MROWS * NOUT * 4;
  float* rg  = (float*)(ws + o); o += (size_t)MROWS * NOUT * 4;
  float* cxg = (float*)(ws + o); o += (size_t)MROWS * NOUT * 4;
  float* h1  = (float*)(ws + o); o += (size_t)MROWS * NOUT * 4;
  float* x2  = (float*)(ws + o); o += (size_t)MROWS * EMBED * 4;
  (void)ws_size; (void)in_sizes; (void)n_in; (void)out_size;

  dim3 blk(256);

  // --- Weight prep: transpose + convert to bf16 (one-time cost, L2-resident)
  {
    dim3 g(NOUT / 32, NIN / 32);
    transpose_cvt<<<g, blk, 0, stream>>>(Wx,  WxT,  NIN, NOUT);
    transpose_cvt<<<g, blk, 0, stream>>>(Wf,  WfT,  NIN, NOUT);
    transpose_cvt<<<g, blk, 0, stream>>>(Wr,  WrT,  NIN, NOUT);
    transpose_cvt<<<g, blk, 0, stream>>>(Wcx, WcxT, NIN, NOUT);
    dim3 gcl(EMBED / 32, NOUT / 32);
    transpose_cvt<<<gcl, blk, 0, stream>>>(Wcl, WclT, NOUT, EMBED);
  }

  // --- Layer 1 ---
  {
    int n = MROWS * NIN;
    cvt_f32_bf16<<<n / 256, blk, 0, stream>>>(xt, Xbf, n);
  }
  {
    dim3 g(MROWS / 128, NOUT / 64);
    wmma_gemm<<<g, blk, 0, stream>>>(Xbf, WxT,  nullptr, xp,  MROWS, NOUT, NIN, 0);
    wmma_gemm<<<g, blk, 0, stream>>>(Xbf, WfT,  bf_,     fg,  MROWS, NOUT, NIN, 1);
    wmma_gemm<<<g, blk, 0, stream>>>(Xbf, WrT,  br,      rg,  MROWS, NOUT, NIN, 1);
    wmma_gemm<<<g, blk, 0, stream>>>(Xbf, WcxT, bcx,     cxg, MROWS, NOUT, NIN, 0);
  }
  sru_scan<<<BN / 256, blk, 0, stream>>>(fg, xp, rg, cxg, ct, h1, nullptr);

  // --- Layer 2 input projection: x2 = h1 @ Wcl + bcl ---
  {
    int n = MROWS * NOUT;
    cvt_f32_bf16<<<n / 256, blk, 0, stream>>>(h1, Hbf, n);
    dim3 g(MROWS / 128, EMBED / 64);
    wmma_gemm<<<g, blk, 0, stream>>>(Hbf, WclT, bcl, x2, MROWS, EMBED, NOUT, 0);
    int n2 = MROWS * EMBED;
    cvt_f32_bf16<<<n2 / 256, blk, 0, stream>>>(x2, X2bf, n2);
  }

  // --- Layer 2 ---
  {
    dim3 g(MROWS / 128, NOUT / 64);
    wmma_gemm<<<g, blk, 0, stream>>>(X2bf, WxT,  nullptr, xp,  MROWS, NOUT, NIN, 0);
    wmma_gemm<<<g, blk, 0, stream>>>(X2bf, WfT,  bf_,     fg,  MROWS, NOUT, NIN, 1);
    wmma_gemm<<<g, blk, 0, stream>>>(X2bf, WrT,  br,      rg,  MROWS, NOUT, NIN, 1);
    wmma_gemm<<<g, blk, 0, stream>>>(X2bf, WcxT, bcx,     cxg, MROWS, NOUT, NIN, 0);
  }
  // h2 -> d_out, c_last of layer 2 -> d_out tail
  sru_scan<<<BN / 256, blk, 0, stream>>>(fg, xp, rg, cxg, ct + (size_t)BN,
                                         out, out + (size_t)MROWS * NOUT);
}
